// ScaleInvariantAttention_25134148616225
// MI455X (gfx1250) — matrix operand order repaired and checked by reading, not compiled, not run on hardware
//
#include <hip/hip_runtime.h>
#include <math.h>

// ---------------------------------------------------------------------------
// ScaleInvariantAttention for MI455X (gfx1250, wave32, WMMA).
// B=2, S=2048, E=1024, H=16, D=64.  All matmuls via v_wmma_f32_16x16x32_bf16.
// Async global->LDS staging (ASYNCcnt) for attention K blocks; double-buffered
// LDS tiles; global_prefetch for the GEMM A stream; WMMA operand-reuse hints.
// ---------------------------------------------------------------------------

typedef __attribute__((ext_vector_type(16))) __bf16 v16bf;
typedef __attribute__((ext_vector_type(8)))  float  v8f;

#define SEQ   2048
#define EMB   1024
#define NH    16
#define HD    64
#define ROWS  4096   // B*S

static __device__ inline unsigned short f32_to_bf16(float f) {
    unsigned int u = __float_as_uint(f);
    u += 0x7FFFu + ((u >> 16) & 1u);       // round-to-nearest-even
    return (unsigned short)(u >> 16);
}

static __device__ inline v16bf as_v16bf(uint4 lo, uint4 hi) {
    union { uint4 q[2]; v16bf v; } u;
    u.q[0] = lo; u.q[1] = hi;
    return u.v;
}

// Low 32 bits of a generic pointer to LDS == LDS byte address (ISA: aperture
// check discards upper bits).
static __device__ inline unsigned lds_off32(const void* p) {
    return (unsigned)(unsigned long long)p;
}

// GLOBAL_LOAD_ASYNC_TO_LDS_B128 (VGLOBAL op 98): vdst = LDS byte address,
// vaddr = 64-bit global address, saddr = off. Tracked with ASYNCcnt.
static __device__ inline void async_copy_b128(unsigned lds_dst, const void* gsrc) {
    asm volatile("global_load_async_to_lds_b128 %0, %1, off"
                 :: "v"(lds_dst), "v"(gsrc)
                 : "memory");
}
static __device__ inline void wait_async0() {
    asm volatile("s_wait_asynccnt 0" ::: "memory");
}
// compiler-only fence: wave-private LDS RAW is ordered in HW (LDS in-order
// per wave), only stop the compiler from reordering.
static __device__ inline void compiler_fence() {
    asm volatile("" ::: "memory");
}

// ---------------------------------------------------------------------------
// fp32 -> bf16 bulk conversion (4 elems / thread)
// ---------------------------------------------------------------------------
__global__ __launch_bounds__(256) void cvt_bf16_kernel(
    const float* __restrict__ in, unsigned short* __restrict__ out, int n) {
    int i = (blockIdx.x * 256 + threadIdx.x) * 4;
    if (i + 3 < n) {
        float4 f = *reinterpret_cast<const float4*>(&in[i]);
        out[i + 0] = f32_to_bf16(f.x);
        out[i + 1] = f32_to_bf16(f.y);
        out[i + 2] = f32_to_bf16(f.z);
        out[i + 3] = f32_to_bf16(f.w);
    }
}

// ---------------------------------------------------------------------------
// Scale-adaptation MLP:  xi -> Linear(1,32) -> LN -> GELU -> Linear(32,64)
// adapt[b,h,d] = 1 + sigmoid(gate[h]) * xe[b,d].   One block, 64 threads.
// ---------------------------------------------------------------------------
__global__ __launch_bounds__(64) void prep_adapt_kernel(
    const float* __restrict__ xi, const float* __restrict__ w1,
    const float* __restrict__ b1, const float* __restrict__ lng,
    const float* __restrict__ lnb, const float* __restrict__ w2,
    const float* __restrict__ b2, const float* __restrict__ gate,
    float* __restrict__ adapt) {
    __shared__ float xg[2][32];
    const int t = threadIdx.x;
    const int b = t >> 5;     // batch = wave
    const int j = t & 31;     // feature 0..31

    float x = xi[b] * w1[j] + b1[j];
    float s = x;
    #pragma unroll
    for (int m = 1; m < 32; m <<= 1) s += __shfl_xor(s, m, 32);
    float mu = s * (1.0f / 32.0f);
    float dv = x - mu;
    float vs = dv * dv;
    #pragma unroll
    for (int m = 1; m < 32; m <<= 1) vs += __shfl_xor(vs, m, 32);
    float xn = dv * rsqrtf(vs * (1.0f / 32.0f) + 1e-5f) * lng[j] + lnb[j];
    xg[b][j] = 0.5f * xn * (1.0f + erff(xn * 0.70710678118654752f));
    __syncthreads();

    for (int idx = t; idx < 2 * HD; idx += 64) {
        int bb = idx >> 6, d = idx & 63;
        float xe = b2[d];
        #pragma unroll 8
        for (int k = 0; k < 32; k++) xe += xg[bb][k] * w2[k * HD + d];
        for (int h = 0; h < NH; h++) {
            float sg = 1.0f / (1.0f + __expf(-gate[h]));
            adapt[(bb * NH + h) * HD + d] = 1.0f + sg * xe;
        }
    }
}

// ---------------------------------------------------------------------------
// Unified WMMA GEMM: out = X[4096,1024] * W[1024,1024] + bias (+adapt scale).
// 256 thr = 8 waves -> 256x64 block tile; each wave owns 32 rows (two A
// fragments, 8 accumulators) so 8 WMMAs amortize one set of 4 B fragments.
// W chunk [32k x 64n] double-buffered in LDS, stored TRANSPOSED so
// B-fragments are contiguous ds_load_b128 pairs (all hoisted before the WMMA
// burst).  reuse_b hint on the 2nd WMMA of each same-B pair.
// ---------------------------------------------------------------------------
__global__ __launch_bounds__(256) void gemm_wmma_kernel(
    const unsigned short* __restrict__ X,   // [4096][1024] bf16
    const unsigned short* __restrict__ W,   // [1024][1024] bf16 (row=k, col=n)
    const float* __restrict__ bias,         // [1024]
    const float* __restrict__ adapt,        // [2][16][64] or nullptr
    unsigned short* __restrict__ outBF,     // [B][H][S][D] or nullptr
    float* __restrict__ outF32) {           // [4096][1024] or nullptr
    __shared__ __attribute__((aligned(16))) unsigned short Wt[2][64][32]; // [n][k]

    const int tid  = threadIdx.x;
    const int wave = tid >> 5;
    const int lane = tid & 31;
    const int half = lane >> 4;
    const int l15  = lane & 15;
    const int m0   = blockIdx.x * 256;
    const int n0   = blockIdx.y * 64;
    const int rbase = m0 + wave * 32;       // 32 rows per wave

    const int kk = tid >> 3;          // 0..31  (k within chunk)
    const int nn = (tid & 7) * 8;     // 0..56  (n within chunk)

    auto ldW = [&](int k0) -> uint4 {
        return *reinterpret_cast<const uint4*>(&W[(size_t)(k0 + kk) * EMB + n0 + nn]);
    };
    auto stW = [&](int bufi, uint4 w) {
        const unsigned short* w16 = reinterpret_cast<const unsigned short*>(&w);
        #pragma unroll
        for (int i = 0; i < 8; i++) Wt[bufi][nn + i][kk] = w16[i];
    };

    v8f acc[2][4];
    #pragma unroll
    for (int mi = 0; mi < 2; mi++)
        #pragma unroll
        for (int t = 0; t < 4; t++)
            #pragma unroll
            for (int i = 0; i < 8; i++) acc[mi][t][i] = 0.0f;

    stW(0, ldW(0));
    __syncthreads();

    int buf = 0;
    for (int k0 = 0; k0 < EMB; k0 += 32, buf ^= 1) {
        const bool more = (k0 + 32) < EMB;
        uint4 wnext = {};
        if (more) wnext = ldW(k0 + 32);        // overlap with compute

        // two A fragments (rows rbase+l15 and rbase+16+l15)
        v16bf afrag[2];
        #pragma unroll
        for (int mi = 0; mi < 2; mi++) {
            const unsigned short* xrow =
                &X[(size_t)(rbase + mi * 16 + l15) * EMB + k0];
            if (more) __builtin_prefetch(xrow + 32, 0, 1);   // global_prefetch_b8
            uint4 a_lo = *reinterpret_cast<const uint4*>(&xrow[16 * half]);
            uint4 a_hi = *reinterpret_cast<const uint4*>(&xrow[16 * half + 8]);
            afrag[mi] = as_v16bf(a_lo, a_hi);
        }

        // hoist all four B fragments (8 ds_load_b128, clauseable)
        v16bf bfrag[4];
        #pragma unroll
        for (int t = 0; t < 4; t++) {
            const uint4* bp = reinterpret_cast<const uint4*>(
                &Wt[buf][t * 16 + l15][16 * half]);
            bfrag[t] = as_v16bf(bp[0], bp[1]);
        }

        // 8 WMMAs; second of each same-B pair carries the reuse_b hint
        #pragma unroll
        for (int t = 0; t < 4; t++) {
            acc[0][t] = __builtin_amdgcn_wmma_f32_16x16x32_bf16(
                false, afrag[0], false, bfrag[t], (short)0, acc[0][t],
                false, false);
            acc[1][t] = __builtin_amdgcn_wmma_f32_16x16x32_bf16(
                false, afrag[1], false, bfrag[t], (short)0, acc[1][t],
                false, true /*reuse_b*/);
        }
        if (more) stW(buf ^ 1, wnext);
        __syncthreads();
    }

    // epilogue: bias (+ adapt) and store
    #pragma unroll
    for (int mi = 0; mi < 2; mi++)
        #pragma unroll
        for (int t = 0; t < 4; t++) {
            const int c = n0 + t * 16 + l15;        // output column 0..1023
            const int h = c >> 6, d = c & 63;
            const float bv = bias[c];
            #pragma unroll
            for (int r = 0; r < 8; r++) {
                const int row = rbase + mi * 16 + half * 8 + r;
                const int b = row >> 11, sidx = row & (SEQ - 1);
                float val = acc[mi][t][r] + bv;
                if (adapt) val *= adapt[((b << 4) + h) * HD + d];
                if (outBF)
                    outBF[(((size_t)((b << 4) + h)) * SEQ + sidx) * HD + d] =
                        f32_to_bf16(val);
                if (outF32)
                    outF32[(size_t)row * EMB + c] = val;
            }
        }
}

// ---------------------------------------------------------------------------
// Flash attention: one (b,h) per blockIdx.y, 128 query rows per blockIdx.x.
// Key blocks of 64 double-buffered in LDS: K staged via async global->LDS
// (ASYNCcnt), V staged transposed via registers. Scores & PV via bf16 WMMA
// with reuse_a hints (same A across the 4 N-subtiles); online softmax in
// registers; P re-shaped through wave-private LDS (per-wave LDS is in-order).
// ---------------------------------------------------------------------------
__global__ __launch_bounds__(256) void attn_wmma_kernel(
    const unsigned short* __restrict__ Q,   // [32][2048][64] bf16
    const unsigned short* __restrict__ K,
    const unsigned short* __restrict__ V,
    unsigned short* __restrict__ O) {       // [4096][1024] bf16
    __shared__ __attribute__((aligned(16))) unsigned short Kblk[2][64][64];
    __shared__ __attribute__((aligned(16))) unsigned short Vt[2][64][64]; // [d][key]
    __shared__ __attribute__((aligned(16))) unsigned short Pbuf[8][16][64];

    const int tid  = threadIdx.x;
    const int wave = tid >> 5;
    const int lane = tid & 31;
    const int half = lane >> 4;
    const int l15  = lane & 15;
    const int bh   = blockIdx.y;                 // b*16 + h
    const int q0   = blockIdx.x * 128 + wave * 16;
    const size_t base = (size_t)bh * SEQ * HD;

    const int srow = tid >> 2;          // 0..63 : staging row (key index)
    const int scol = (tid & 3) * 16;    // 0..48 : staging col base (d index)

    auto stageK_async = [&](int kb, int pb) {
        const size_t goff = base + (size_t)(kb * 64 + srow) * HD + scol;
        async_copy_b128(lds_off32(&Kblk[pb][srow][scol]),     &K[goff]);
        async_copy_b128(lds_off32(&Kblk[pb][srow][scol + 8]), &K[goff + 8]);
    };
    auto loadV = [&](int kb, uint4& v0, uint4& v1) {
        const size_t goff = base + (size_t)(kb * 64 + srow) * HD + scol;
        const uint4* vp = reinterpret_cast<const uint4*>(&V[goff]);
        v0 = vp[0];
        v1 = vp[1];
    };
    auto scatterV = [&](int pb, uint4 v0, uint4 v1) {
        const unsigned short* e0 = reinterpret_cast<const unsigned short*>(&v0);
        const unsigned short* e1 = reinterpret_cast<const unsigned short*>(&v1);
        #pragma unroll
        for (int i = 0; i < 8; i++) Vt[pb][scol + i][srow] = e0[i];
        #pragma unroll
        for (int i = 0; i < 8; i++) Vt[pb][scol + 8 + i][srow] = e1[i];
    };

    // preload Q A-fragments for both d-halves (K-steps)
    v16bf qf[2];
    {
        const unsigned short* qrow = &Q[base + (size_t)(q0 + l15) * HD];
        #pragma unroll
        for (int ks = 0; ks < 2; ks++) {
            uint4 lo = *reinterpret_cast<const uint4*>(&qrow[32 * ks + 16 * half]);
            uint4 hi = *reinterpret_cast<const uint4*>(&qrow[32 * ks + 16 * half + 8]);
            qf[ks] = as_v16bf(lo, hi);
        }
    }

    float m_run[8], l_run[8];
    v8f oacc[4];
    #pragma unroll
    for (int r = 0; r < 8; r++) { m_run[r] = -1e30f; l_run[r] = 0.0f; }
    #pragma unroll
    for (int t = 0; t < 4; t++)
        #pragma unroll
        for (int i = 0; i < 8; i++) oacc[t][i] = 0.0f;

    // prologue: stage key block 0 into buffer 0
    stageK_async(0, 0);
    {
        uint4 v0, v1;
        loadV(0, v0, v1);
        scatterV(0, v0, v1);
    }
    wait_async0();
    __syncthreads();

    int pb = 0;
    for (int kb = 0; kb < SEQ / 64; kb++, pb ^= 1) {
        const bool more = (kb + 1) < SEQ / 64;
        uint4 nv0 = {}, nv1 = {};
        if (more) {
            stageK_async(kb + 1, pb ^ 1);   // async into other buffer
            loadV(kb + 1, nv0, nv1);        // V to registers, scatter later
        }

        // ---- scores: S = Q * K^T * (1/sqrt(64)) ----
        v8f sacc[4];
        #pragma unroll
        for (int t = 0; t < 4; t++)
            #pragma unroll
            for (int i = 0; i < 8; i++) sacc[t][i] = 0.0f;

        #pragma unroll
        for (int ks = 0; ks < 2; ks++) {
            v16bf bfrag[4];
            #pragma unroll
            for (int t = 0; t < 4; t++) {
                const uint4* bp = reinterpret_cast<const uint4*>(
                    &Kblk[pb][t * 16 + l15][32 * ks + 16 * half]);
                bfrag[t] = as_v16bf(bp[0], bp[1]);
            }
            // t = 0 peeled (reuse_a must be an immediate constant)
            sacc[0] = __builtin_amdgcn_wmma_f32_16x16x32_bf16(
                false, qf[ks], false, bfrag[0], (short)0, sacc[0],
                false, false);
            #pragma unroll
            for (int t = 1; t < 4; t++)
                sacc[t] = __builtin_amdgcn_wmma_f32_16x16x32_bf16(
                    false, qf[ks], false, bfrag[t], (short)0, sacc[t],
                    true /*reuse_a*/, false);
        }

        // ---- online softmax (rows m = half*8 + r, cols across 16 lanes) ----
        #pragma unroll
        for (int r = 0; r < 8; r++) {
            float mx = sacc[0][r] * 0.125f;
            #pragma unroll
            for (int t = 0; t < 4; t++) {
                sacc[t][r] *= 0.125f;
                mx = fmaxf(mx, sacc[t][r]);
            }
            #pragma unroll
            for (int m = 1; m < 16; m <<= 1) mx = fmaxf(mx, __shfl_xor(mx, m, 32));
            const float mnew = fmaxf(m_run[r], mx);
            const float corr = __expf(m_run[r] - mnew);
            m_run[r] = mnew;
            float rsum = 0.0f;
            #pragma unroll
            for (int t = 0; t < 4; t++) {
                float p = __expf(sacc[t][r] - mnew);
                sacc[t][r] = p;
                rsum += p;
            }
            #pragma unroll
            for (int m = 1; m < 16; m <<= 1) rsum += __shfl_xor(rsum, m, 32);
            l_run[r] = l_run[r] * corr + rsum;
            #pragma unroll
            for (int t = 0; t < 4; t++) oacc[t][r] *= corr;
        }

        // ---- spill P (bf16) to wave-private LDS, row-major [16][64] ----
        #pragma unroll
        for (int t = 0; t < 4; t++)
            #pragma unroll
            for (int r = 0; r < 8; r++)
                Pbuf[wave][half * 8 + r][t * 16 + l15] = f32_to_bf16(sacc[t][r]);
        compiler_fence();   // HW keeps per-wave LDS ops in order

        // ---- O += P * V  (A = P from LDS, B = Vt contiguous) ----
        #pragma unroll
        for (int ks = 0; ks < 2; ks++) {
            const unsigned short* prow = &Pbuf[wave][l15][32 * ks];
            uint4 alo = *reinterpret_cast<const uint4*>(&prow[16 * half]);
            uint4 ahi = *reinterpret_cast<const uint4*>(&prow[16 * half + 8]);
            v16bf pa = as_v16bf(alo, ahi);
            v16bf vfrag[4];
            #pragma unroll
            for (int t = 0; t < 4; t++) {
                const uint4* bp = reinterpret_cast<const uint4*>(
                    &Vt[pb][t * 16 + l15][32 * ks + 16 * half]);
                vfrag[t] = as_v16bf(bp[0], bp[1]);
            }
            // t = 0 peeled (reuse_a must be an immediate constant)
            oacc[0] = __builtin_amdgcn_wmma_f32_16x16x32_bf16(
                false, pa, false, vfrag[0], (short)0, oacc[0],
                false, false);
            #pragma unroll
            for (int t = 1; t < 4; t++)
                oacc[t] = __builtin_amdgcn_wmma_f32_16x16x32_bf16(
                    false, pa, false, vfrag[t], (short)0, oacc[t],
                    true /*reuse_a*/, false);
        }

        if (more) {
            scatterV(pb ^ 1, nv0, nv1);     // fill other V buffer
            wait_async0();                  // own async K copies landed
        }
        __syncthreads();
    }

    // ---- normalize and store to O in [B,S,E] bf16 layout ----
    const int h = bh & 15, b = bh >> 4;
    #pragma unroll
    for (int t = 0; t < 4; t++)
        #pragma unroll
        for (int r = 0; r < 8; r++) {
            const int sidx = q0 + half * 8 + r;
            const float val = oacc[t][r] / l_run[r];
            O[((size_t)(b * SEQ + sidx)) * EMB + h * HD + t * 16 + l15] =
                f32_to_bf16(val);
        }
}

// ---------------------------------------------------------------------------
extern "C" void kernel_launch(void* const* d_in, const int* in_sizes, int n_in,
                              void* d_out, int out_size, void* d_ws, size_t ws_size,
                              hipStream_t stream) {
    (void)in_sizes; (void)n_in; (void)out_size; (void)ws_size;
    const float* query = (const float*)d_in[0];
    const float* xi    = (const float*)d_in[1];
    const float* Wq = (const float*)d_in[2];  const float* bq = (const float*)d_in[3];
    const float* Wk = (const float*)d_in[4];  const float* bk = (const float*)d_in[5];
    const float* Wv = (const float*)d_in[6];  const float* bv = (const float*)d_in[7];
    const float* Wo = (const float*)d_in[8];  const float* bo = (const float*)d_in[9];
    const float* ew1 = (const float*)d_in[10]; const float* eb1 = (const float*)d_in[11];
    const float* lng = (const float*)d_in[12]; const float* lnb = (const float*)d_in[13];
    const float* ew2 = (const float*)d_in[14]; const float* eb2 = (const float*)d_in[15];
    const float* gate = (const float*)d_in[16];
    float* out = (float*)d_out;

    // workspace carve-up (all 256B aligned)
    char* ws = (char*)d_ws;
    size_t off = 0;
    auto carve = [&](size_t bytes) -> void* {
        void* p = ws + off;
        off += (bytes + 255) & ~(size_t)255;
        return p;
    };
    unsigned short* Xbf  = (unsigned short*)carve((size_t)ROWS * EMB * 2);
    unsigned short* Wqbf = (unsigned short*)carve((size_t)EMB * EMB * 2);
    unsigned short* Wkbf = (unsigned short*)carve((size_t)EMB * EMB * 2);
    unsigned short* Wvbf = (unsigned short*)carve((size_t)EMB * EMB * 2);
    unsigned short* Wobf = (unsigned short*)carve((size_t)EMB * EMB * 2);
    unsigned short* Qbf  = (unsigned short*)carve((size_t)ROWS * EMB * 2);
    unsigned short* Kbf  = (unsigned short*)carve((size_t)ROWS * EMB * 2);
    unsigned short* Vbf  = (unsigned short*)carve((size_t)ROWS * EMB * 2);
    unsigned short* Obf  = (unsigned short*)carve((size_t)ROWS * EMB * 2);
    float* adapt = (float*)carve((size_t)2 * NH * HD * 4);

    // 1) scale-adaptation MLP
    prep_adapt_kernel<<<1, 64, 0, stream>>>(xi, ew1, eb1, lng, lnb, ew2, eb2,
                                            gate, adapt);
    // 2) bf16 conversions
    cvt_bf16_kernel<<<(ROWS * EMB) / 1024, 256, 0, stream>>>(query, Xbf, ROWS * EMB);
    cvt_bf16_kernel<<<(EMB * EMB) / 1024, 256, 0, stream>>>(Wq, Wqbf, EMB * EMB);
    cvt_bf16_kernel<<<(EMB * EMB) / 1024, 256, 0, stream>>>(Wk, Wkbf, EMB * EMB);
    cvt_bf16_kernel<<<(EMB * EMB) / 1024, 256, 0, stream>>>(Wv, Wvbf, EMB * EMB);
    cvt_bf16_kernel<<<(EMB * EMB) / 1024, 256, 0, stream>>>(Wo, Wobf, EMB * EMB);

    // 3) projections (Q,K get adapt scaling fused into the GEMM epilogue)
    dim3 gg(ROWS / 256, EMB / 64);
    gemm_wmma_kernel<<<gg, 256, 0, stream>>>(Xbf, Wqbf, bq, adapt, Qbf, nullptr);
    gemm_wmma_kernel<<<gg, 256, 0, stream>>>(Xbf, Wkbf, bk, adapt, Kbf, nullptr);
    gemm_wmma_kernel<<<gg, 256, 0, stream>>>(Xbf, Wvbf, bv, nullptr, Vbf, nullptr);

    // 4) flash attention (16 query blocks x 32 (b,h) pairs)
    attn_wmma_kernel<<<dim3(SEQ / 128, 2 * NH), 256, 0, stream>>>(Qbf, Kbf, Vbf, Obf);

    // 5) output projection, fp32 result straight to d_out
    gemm_wmma_kernel<<<gg, 256, 0, stream>>>(Obf, Wobf, bo, nullptr, nullptr, out);
}
